// SelfAttention_21285857919299
// MI455X (gfx1250) — matrix operand order repaired and checked by reading, not compiled, not run on hardware
//
#include <hip/hip_runtime.h>
#include <stdint.h>

#define B_ 2
#define H_ 16
#define T_ 2048
#define D_ 64

#define TQ 128   // query rows per workgroup (8 waves x 16 rows)
#define TK 64    // key block size
#define KROW 72  // padded LDS row stride in bf16 elements (144 B -> conflict-free)

typedef __attribute__((ext_vector_type(16))) __bf16 v16bf;
typedef __attribute__((ext_vector_type(8)))  float  v8f;

#define LOG2E 1.4426950408889634f
#define FILL2 (1e-10f * LOG2E)   // masked_fill value in log2 domain

union V16 {
  v16bf bf;
  uint4 q[2];
};

struct __align__(16) SMem {
  __bf16 Ks[TK * KROW];      // K tile as [key][d] (bf16)
  __bf16 Vs[D_ * KROW];      // V tile as [d][key] (bf16)
  __bf16 Ps[8 * 16 * KROW];  // per-wave P scratch, row-major [row][key]
};

__device__ __forceinline__ uint32_t pack_bf16(float a, float b) {
  union { __bf16 h[2]; uint32_t u; } r;
  r.h[0] = (__bf16)a;
  r.h[1] = (__bf16)b;
  return r.u;
}

__device__ __forceinline__ void load8_scaled_bf16(const float* __restrict__ p,
                                                  float sc, v16bf& dst, int o) {
  float4 a = ((const float4*)p)[0];
  float4 b = ((const float4*)p)[1];
  dst[o + 0] = (__bf16)(a.x * sc); dst[o + 1] = (__bf16)(a.y * sc);
  dst[o + 2] = (__bf16)(a.z * sc); dst[o + 3] = (__bf16)(a.w * sc);
  dst[o + 4] = (__bf16)(b.x * sc); dst[o + 5] = (__bf16)(b.y * sc);
  dst[o + 6] = (__bf16)(b.z * sc); dst[o + 7] = (__bf16)(b.w * sc);
}

// Per-wave compute for one 64-key block. CAUSAL=false for fully-interior blocks.
template <bool CAUSAL>
__device__ __forceinline__ void attn_block(
    SMem& sm, const v16bf (&qa)[2], const float (&hm)[8],
    float (&m)[8], float (&l)[8], v8f (&acc)[4],
    __bf16* __restrict__ pw, const float* __restrict__ vmask_row,
    int kb0, int qrow_base, int half, int l16)
{
  const v8f vzero = {0.f, 0.f, 0.f, 0.f, 0.f, 0.f, 0.f, 0.f};

  // ---- S = Q * K^T (log2-scaled): 4 column tiles, K-dim = D in 2 chunks of 32
  v8f st[4];
  #pragma unroll
  for (int nt = 0; nt < 4; ++nt) {
    st[nt] = vzero;
    const int key = nt * 16 + l16;            // B layout: column per lane
    #pragma unroll
    for (int c = 0; c < 2; ++c) {
      V16 kbv;
      const int off = key * KROW + c * 32 + half * 8;  // K in table order (8h, 16+8h)
      kbv.q[0] = *(const uint4*)&sm.Ks[off];
      kbv.q[1] = *(const uint4*)&sm.Ks[off + 16];
      st[nt] = __builtin_amdgcn_wmma_f32_16x16x32_bf16(
          false, qa[c], false, kbv.bf, (short)0, st[nt], false, false);
    }
  }

  // ---- masks + online softmax (C layout: col = l16 per tile, row = slot)
  float vm[4];
  #pragma unroll
  for (int nt = 0; nt < 4; ++nt)
    vm[nt] = vmask_row[kb0 + nt * 16 + l16];

  #pragma unroll
  for (int j = 0; j < 8; ++j) {
    const int rowg = qrow_base + half * 8 + j;
    float mx = -__builtin_inff();
    #pragma unroll
    for (int nt = 0; nt < 4; ++nt) {
      float s = st[nt][j] * (vm[nt] * hm[j]);
      s = (s == 0.f) ? FILL2 : s;             // masked_fill(s==0, 1e-10), log2 dom
      if (CAUSAL) {
        const int colg = kb0 + nt * 16 + l16;
        if (colg > rowg) s = -__builtin_inff();
      }
      st[nt][j] = s;
      mx = fmaxf(mx, s);
    }
    #pragma unroll
    for (int x = 1; x < 16; x <<= 1)
      mx = fmaxf(mx, __shfl_xor(mx, x, 32));  // row-max across 16-lane half
    const float mn = fmaxf(m[j], mx);
    const float sc = __builtin_amdgcn_exp2f(m[j] - mn);
    m[j] = mn;
    float rs = 0.f;
    #pragma unroll
    for (int nt = 0; nt < 4; ++nt) {
      float p = __builtin_amdgcn_exp2f(st[nt][j] - mn);
      st[nt][j] = p;
      rs += p;
    }
    #pragma unroll
    for (int x = 1; x < 16; x <<= 1)
      rs += __shfl_xor(rs, x, 32);
    l[j] = l[j] * sc + rs;
    #pragma unroll
    for (int dt = 0; dt < 4; ++dt)
      acc[dt][j] *= sc;                        // rescale running O
  }

  // ---- intra-wave transpose of P: C layout -> row-major LDS scratch (bf16)
  #pragma unroll
  for (int j = 0; j < 8; ++j)
    #pragma unroll
    for (int nt = 0; nt < 4; ++nt)
      pw[(half * 8 + j) * KROW + nt * 16 + l16] = (__bf16)st[nt][j];
  asm volatile("s_wait_dscnt 0" ::: "memory"); // DS RAW within wave
  __builtin_amdgcn_wave_barrier();

  // ---- O += P * V : A = P (16x32 keys), B = V (32 keys x 16 d), 4 d-tiles
  #pragma unroll
  for (int c = 0; c < 2; ++c) {
    const int kc = c * 32;
    V16 pa;
    const int pb = l16 * KROW + kc + half * 8;  // A layout from scratch
    pa.q[0] = *(const uint4*)&pw[pb];
    pa.q[1] = *(const uint4*)&pw[pb + 16];
    #pragma unroll
    for (int dt = 0; dt < 4; ++dt) {
      V16 vb;
      const int vo = (dt * 16 + l16) * KROW + kc + half * 8;
      vb.q[0] = *(const uint4*)&sm.Vs[vo];
      vb.q[1] = *(const uint4*)&sm.Vs[vo + 16];
      acc[dt] = __builtin_amdgcn_wmma_f32_16x16x32_bf16(
          false, pa.bf, false, vb.bf, (short)0, acc[dt], false, false);
    }
  }
}

__global__ __launch_bounds__(256, 1) void fa_fwd(
    const float* __restrict__ q, const float* __restrict__ k,
    const float* __restrict__ v, const float* __restrict__ f,
    const float* __restrict__ vmask, const float* __restrict__ hmask,
    float* __restrict__ out)
{
  __shared__ SMem sm;
  const int tid  = threadIdx.x;
  const int lane = tid & 31;
  const int wave = tid >> 5;
  const int half = lane >> 4;   // which 16-lane half of the wave
  const int l16  = lane & 15;
  const int q0   = blockIdx.x * TQ;
  const int bh   = blockIdx.y;
  const int b    = bh / H_;

  const float* qp = q + (size_t)bh * T_ * D_;   // [T][D]
  const float* kp = k + (size_t)bh * D_ * T_;   // [D][T] (pre-transposed)
  const float* vp = v + (size_t)bh * T_ * D_;   // [T][D]
  float*       op = out + (size_t)bh * T_ * D_;
  const float* vmask_row = vmask + b * T_;
  const float qscale = LOG2E / f[0];  // fold 1/sqrt(D) and log2(e) into Q

  const int qrow_base = q0 + wave * 16;

  // ---- Q in WMMA A layout (16x32 bf16 per chunk), pre-scaled
  const int arow = qrow_base + l16;
  v16bf qa[2];
  #pragma unroll
  for (int c = 0; c < 2; ++c) {
    const float* qr = qp + (size_t)arow * D_ + c * 32 + half * 8;
    load8_scaled_bf16(qr,      qscale, qa[c], 0);
    load8_scaled_bf16(qr + 16, qscale, qa[c], 8);
  }

  // per-slot row mask (C layout: slot j holds row j or j+8 depending on half)
  float hm[8];
  #pragma unroll
  for (int j = 0; j < 8; ++j)
    hm[j] = hmask[b * T_ + qrow_base + half * 8 + j];

  float m[8], l[8];
  const v8f vzero = {0.f, 0.f, 0.f, 0.f, 0.f, 0.f, 0.f, 0.f};
  v8f acc[4];
  #pragma unroll
  for (int j = 0; j < 8; ++j) { m[j] = -__builtin_inff(); l[j] = 0.f; }
  #pragma unroll
  for (int t = 0; t < 4; ++t) acc[t] = vzero;

  __bf16* pw = &sm.Ps[wave * 16 * KROW];
  const int nkb = (q0 + TQ) / TK;   // causal: key blocks up to end of query tile

  for (int kb = 0; kb < nkb; ++kb) {
    const int kb0 = kb * TK;
    __syncthreads();

    // ---- staging: K (64 keys x 64 d) -> LDS [key][d], packed-pair bf16 stores
    #pragma unroll
    for (int r = 0; r < 8; ++r) {
      int idx = r * 256 + tid;            // 0..2047 pairs
      int dp = idx >> 6, t = idx & 63;    // dp = d-pair, coalesced along t
      float a  = kp[(size_t)(2 * dp)     * T_ + kb0 + t];
      float b2 = kp[(size_t)(2 * dp + 1) * T_ + kb0 + t];
      *(uint32_t*)&sm.Ks[t * KROW + 2 * dp] = pack_bf16(a, b2);
    }
    // ---- V (64 keys x 64 d) -> LDS [d][key] (transpose), packed-pair stores
    #pragma unroll
    for (int r = 0; r < 8; ++r) {
      int idx = r * 256 + tid;
      int tp = idx >> 6, d = idx & 63;    // tp = key-pair, coalesced along d
      float a  = vp[(size_t)(kb0 + 2 * tp)     * D_ + d];
      float b2 = vp[(size_t)(kb0 + 2 * tp + 1) * D_ + d];
      *(uint32_t*)&sm.Vs[d * KROW + 2 * tp] = pack_bf16(a, b2);
    }
    __syncthreads();

    // prefetch next key block while this one is computed
    if (kb + 1 < nkb) {
      const int t = tid & 63, d4 = (tid >> 6) * 16;
      __builtin_prefetch(&kp[(size_t)d4 * T_ + kb0 + TK + t], 0, 0);
      __builtin_prefetch(&vp[(size_t)(kb0 + TK + t) * D_ + d4], 0, 0);
    }

    // interior blocks skip the causal cmp/cndmask entirely (wave-uniform)
    if (kb0 + TK - 1 <= qrow_base)
      attn_block<false>(sm, qa, hm, m, l, acc, pw, vmask_row,
                        kb0, qrow_base, half, l16);
    else
      attn_block<true>(sm, qa, hm, m, l, acc, pw, vmask_row,
                       kb0, qrow_base, half, l16);
  }

  // ---- epilogue: normalize by row sum (fast rcp), store f32
  float rl[8];
  #pragma unroll
  for (int j = 0; j < 8; ++j)
    rl[j] = __builtin_amdgcn_rcpf(l[j]);
  #pragma unroll
  for (int dt = 0; dt < 4; ++dt)
    #pragma unroll
    for (int j = 0; j < 8; ++j) {
      const int rowg = qrow_base + half * 8 + j;
      op[(size_t)rowg * D_ + dt * 16 + l16] = acc[dt][j] * rl[j];
    }
}

extern "C" void kernel_launch(void* const* d_in, const int* in_sizes, int n_in,
                              void* d_out, int out_size, void* d_ws, size_t ws_size,
                              hipStream_t stream) {
  (void)in_sizes; (void)n_in; (void)out_size; (void)d_ws; (void)ws_size;
  const float* q     = (const float*)d_in[0];
  const float* k     = (const float*)d_in[1];
  const float* v     = (const float*)d_in[2];
  const float* f     = (const float*)d_in[3];
  const float* vmask = (const float*)d_in[4];
  const float* hmask = (const float*)d_in[5];
  float* out = (float*)d_out;

  dim3 grid(T_ / TQ, B_ * H_);   // 16 query tiles x 32 (b,h) pairs
  fa_fwd<<<grid, 256, 0, stream>>>(q, k, v, f, vmask, hmask, out);
}
